// KMeansInference_49770081026629
// MI455X (gfx1250) — matrix operand order repaired and checked by reading, not compiled, not run on hardware
//
#include <hip/hip_runtime.h>
#include <hip/hip_bf16.h>
#include <stdint.h>

// ---------------------------------------------------------------------------
// KMeans assignment on gfx1250:  argmin_k ( ||c_k||^2 - 2 * x . c_k )
// GEMM [N x D] @ [D x Kpad] with bf16 hi/lo split (3x v_wmma_f32_16x16x32_bf16)
// ---------------------------------------------------------------------------

typedef __attribute__((ext_vector_type(16))) __bf16        v16bf;
typedef __attribute__((ext_vector_type(8)))  float         v8f;
typedef __attribute__((ext_vector_type(4)))  unsigned int  u32x4;

#define N_PTS   262144
#define K_CLUS  500
#define K_PAD   512
#define DIM     768
#define NT      (K_PAD / 16)   // 32 cluster tiles of 16
#define KC      (DIM / 32)     // 24 k-chunks of 32
#define ROWS    64             // rows per block
#define THREADS 256            // 8 waves

// Workspace layout in dwords:
#define WS_BHI  0                       // NT*KC*32*8 = 196608 dwords (B hi frags)
#define WS_BLO  (NT * KC * 32 * 8)      // 196608 dwords (B lo frags)
#define WS_CSQ  (2 * NT * KC * 32 * 8)  // 512 dwords (||c||^2, padded)

// A region in LDS: 2 matrices (hi, lo), each 4 mtiles * 24 kc * 32 lanes * 8 dwords
#define LDS_A_DWORDS   (4 * KC * 32 * 8)        // 24576 per matrix
#define LDS_COMB_BASE  (2 * LDS_A_DWORDS)       // 49152
#define LDS_TOTAL_B    ((LDS_COMB_BASE + 256) * 4)  // 197632 bytes

__device__ __forceinline__ unsigned int f32_to_bf16_rne(float x) {
  unsigned int u = __float_as_uint(x);
  unsigned int r = u + 0x7FFFu + ((u >> 16) & 1u);
  return r >> 16;
}

__device__ __forceinline__ void split_bf16(float x, unsigned int& hi, unsigned int& lo) {
  hi = f32_to_bf16_rne(x);
  float hf = __uint_as_float(hi << 16);
  lo = f32_to_bf16_rne(x - hf);
}

// --------------------------- prep: ||c||^2 ---------------------------------
__global__ void kmeans_prep_csq(const float* __restrict__ C, float* __restrict__ csq) {
  int c = blockIdx.x * blockDim.x + threadIdx.x;
  if (c >= K_PAD) return;
  if (c < K_CLUS) {
    const float* p = C + (size_t)c * DIM;
    float s = 0.f;
    for (int k = 0; k < DIM; ++k) s += p[k] * p[k];
    csq[c] = s;
  } else {
    csq[c] = 3.0e38f;  // padded clusters never win
  }
}

// ---------------- prep: centers -> bf16 hi/lo B-fragments ------------------
// B-matrix (32x16 bf16, wave32) lane layout used here (contiguous k, same order as A):
//   lanes 0-15 : column n = lane,    K = 0..15  packed sequentially in V0..V7
//   lanes 16-31: column n = lane-16, K = 16..31 packed sequentially in V0..V7
// Fragment storage: frag[nt][kc][lane][8 dwords] so the GEMM does 2x b128 loads.
__global__ void kmeans_prep_bfrag(const float* __restrict__ C,
                                  unsigned int* __restrict__ bhi,
                                  unsigned int* __restrict__ blo) {
  int idx = blockIdx.x * blockDim.x + threadIdx.x;
  if (idx >= NT * KC * 32 * 8) return;
  int p    = idx & 7;
  int lane = (idx >> 3) & 31;
  int t    = idx >> 8;          // nt*KC + kc
  int kc   = t % KC;
  int nt   = t / KC;
  int col  = lane & 15;
  int clus = nt * 16 + col;
  int kk   = ((lane >> 4) << 4) | (p << 1);
  int k    = kc * 32 + kk;
  float x0 = 0.f, x1 = 0.f;
  if (clus < K_CLUS) {
    x0 = C[(size_t)clus * DIM + k];
    x1 = C[(size_t)clus * DIM + k + 1];
  }
  unsigned int h0, l0, h1, l1;
  split_bf16(x0, h0, l0);
  split_bf16(x1, h1, l1);
  bhi[idx] = h0 | (h1 << 16);
  blo[idx] = l0 | (l1 << 16);
}

// ------------------------------ main kernel --------------------------------
union Frag {
  u32x4 u[2];
  v16bf v;
};

__global__ __launch_bounds__(THREADS) void kmeans_assign(
    const float* __restrict__ X,
    const unsigned int* __restrict__ bhi,
    const unsigned int* __restrict__ blo,
    const float* __restrict__ csq,
    float* __restrict__ out) {
  extern __shared__ unsigned int smem[];  // [LDS_TOTAL_B/4]

  const int tid  = threadIdx.x;
  const int lane = tid & 31;
  const int wave = tid >> 5;
  const int mtile = wave & 3;    // which 16-row tile of the 64-row block
  const int nhalf = wave >> 2;   // 0: nt 0..15, 1: nt 16..31
  const size_t rowbase = (size_t)blockIdx.x * ROWS;

  // ---- Phase 1: stream 64x768 f32 features, split to bf16 hi/lo, store in
  //      LDS in A-fragment layout (16x32 bf16 A, contiguous k, same order as B frags):
  //      lanes 0-15 : row=lane,    K = 0..15  packed sequentially in V0..V7
  //      lanes 16-31: row=lane-16, K = 16..31 packed sequentially in V0..V7
  for (int i = tid; i < ROWS * DIM / 2; i += THREADS) {
    int row = i / (DIM / 2);
    int k   = (i % (DIM / 2)) * 2;
    const float* src = X + (rowbase + row) * (size_t)DIM + k;
    float x0 = src[0];
    float x1 = src[1];
    unsigned int h0, l0, h1, l1;
    split_bf16(x0, h0, l0);
    split_bf16(x1, h1, l1);
    int mt = row >> 4, rr = row & 15;
    int kc = k >> 5,  kk = k & 31;
    int lane_t = rr + ((kk & 16) ? 16 : 0);
    int p = (kk & 15) >> 1;
    int daddr = ((mt * KC + kc) * 32 + lane_t) * 8 + p;
    smem[daddr]                = h0 | (h1 << 16);
    smem[LDS_A_DWORDS + daddr] = l0 | (l1 << 16);
  }
  __syncthreads();

  // ---- Phase 2: WMMA over 16 n-tiles x 24 k-chunks, bf16 hi/lo x3
  float bestv[8];
  int   besti[8];
#pragma unroll
  for (int r = 0; r < 8; ++r) { bestv[r] = __builtin_inff(); besti[r] = 0; }

  const int cid = ((nhalf ? 16 : 0)) * 16;  // base handled per-nt below
  (void)cid;

  for (int nt = nhalf * 16; nt < nhalf * 16 + 16; ++nt) {
    v8f acc = {0.f, 0.f, 0.f, 0.f, 0.f, 0.f, 0.f, 0.f};
#pragma unroll 4
    for (int kc = 0; kc < KC; ++kc) {
      int abase = ((mtile * KC + kc) * 32 + lane) * 8;
      Frag ah, al, bh, bl;
      const u32x4* ap  = (const u32x4*)(smem + abase);
      const u32x4* alp = (const u32x4*)(smem + LDS_A_DWORDS + abase);
      ah.u[0] = ap[0];  ah.u[1] = ap[1];
      al.u[0] = alp[0]; al.u[1] = alp[1];
      int bbase = ((nt * KC + kc) * 32 + lane) * 8;
      const u32x4* bp  = (const u32x4*)(bhi + bbase);
      const u32x4* blp = (const u32x4*)(blo + bbase);
      bh.u[0] = bp[0];  bh.u[1] = bp[1];
      bl.u[0] = blp[0]; bl.u[1] = blp[1];
      // x.c = hi*hi + hi*lo + lo*hi  (lo*lo dropped, ~2^-16 rel err)
      acc = __builtin_amdgcn_wmma_f32_16x16x32_bf16(false, ah.v, false, bh.v,
                                                    (short)0, acc, false, false);
      acc = __builtin_amdgcn_wmma_f32_16x16x32_bf16(false, ah.v, false, bl.v,
                                                    (short)0, acc, false, false);
      acc = __builtin_amdgcn_wmma_f32_16x16x32_bf16(false, al.v, false, bh.v,
                                                    (short)0, acc, false, false);
    }
    // score = ||c||^2 - 2 x.c ; C/D layout: lanes 0-15 -> M=r, lanes16-31 -> M=8+r
    int cluster = nt * 16 + (lane & 15);
    float cs = csq[cluster];
#pragma unroll
    for (int r = 0; r < 8; ++r) {
      float s = cs - 2.0f * acc[r];
      if (s < bestv[r]) { bestv[r] = s; besti[r] = cluster; }  // nt ascending -> first-idx tie-break
    }
  }

  // ---- Cross-lane argmin within each 16-lane column group
  float* comb_v = (float*)(smem + LDS_COMB_BASE);        // [2][64]
  int*   comb_i = (int*)(smem + LDS_COMB_BASE + 128);    // [2][64]
#pragma unroll
  for (int r = 0; r < 8; ++r) {
    float v = bestv[r];
    int   ix = besti[r];
#pragma unroll
    for (int off = 8; off >= 1; off >>= 1) {  // stays within each 16-lane half
      float ov = __shfl_xor(v, off, 32);
      int   oi = __shfl_xor(ix, off, 32);
      if (ov < v || (ov == v && oi < ix)) { v = ov; ix = oi; }
    }
    if ((lane & 15) == 0) {
      int row = mtile * 16 + ((lane >> 4) << 3) + r;  // lane0: M=r, lane16: M=8+r
      comb_v[nhalf * ROWS + row] = v;
      comb_i[nhalf * ROWS + row] = ix;
    }
  }
  __syncthreads();

  // ---- Combine the two n-halves, write float-encoded indices
  if (tid < ROWS) {
    float v0 = comb_v[tid],        v1 = comb_v[ROWS + tid];
    int   i0 = comb_i[tid],        i1 = comb_i[ROWS + tid];
    int ix = (v1 < v0 || (v1 == v0 && i1 < i0)) ? i1 : i0;
    out[rowbase + tid] = (float)ix;
  }
}

// ------------------------------ launcher -----------------------------------
extern "C" void kernel_launch(void* const* d_in, const int* in_sizes, int n_in,
                              void* d_out, int out_size, void* d_ws, size_t ws_size,
                              hipStream_t stream) {
  const float* X = (const float*)d_in[0];   // features [N, D] f32
  const float* C = (const float*)d_in[1];   // centers  [K, D] f32
  unsigned int* ws = (unsigned int*)d_ws;   // needs ~1.58 MB
  float* out = (float*)d_out;               // [N] indices (float-encoded)

  kmeans_prep_csq<<<(K_PAD + 255) / 256, 256, 0, stream>>>(
      C, (float*)(ws + WS_CSQ));
  kmeans_prep_bfrag<<<(NT * KC * 32 * 8 + 255) / 256, 256, 0, stream>>>(
      C, ws + WS_BHI, ws + WS_BLO);
  kmeans_assign<<<N_PTS / ROWS, THREADS, LDS_TOTAL_B, stream>>>(
      X, ws + WS_BHI, ws + WS_BLO, (const float*)(ws + WS_CSQ), out);
}
